// VisionTransformerMoE_63952063037470
// MI455X (gfx1250) — compile-verified, matrix operand order checked by reading, not run on hardware
//
#include <hip/hip_runtime.h>

#define MPAD 832          // 788 real rows (4*197) padded to 13*64
#define MREAL 788
#define NGEMM_TILE 64

typedef __attribute__((ext_vector_type(16))) _Float16 v16h;
typedef __attribute__((ext_vector_type(8)))  float    v8f;

// ---------------------------------------------------------------------------
// Fragment builders: load fp32, convert to f16 in-register.
// 16-bit A-matrix 16x32 layout (ISA 7.12.2): lane l in [0,15] -> row M = l,
// half = lane>>4 selects K offset: elems 0..7 = K = kb..kb+7, elems 8..15 =
// K = kb+16..kb+23, where kb = kstep + 8*half. B (32x16) assumed symmetric
// with N on lanes.
// ---------------------------------------------------------------------------
__device__ __forceinline__ v16h make_frag(const float* p0, const float* p1) {
  const float4 q0 = *(const float4*)(p0);
  const float4 q1 = *(const float4*)(p0 + 4);
  const float4 q2 = *(const float4*)(p1);
  const float4 q3 = *(const float4*)(p1 + 4);
  v16h f;
  f[0]  = (_Float16)q0.x; f[1]  = (_Float16)q0.y; f[2]  = (_Float16)q0.z; f[3]  = (_Float16)q0.w;
  f[4]  = (_Float16)q1.x; f[5]  = (_Float16)q1.y; f[6]  = (_Float16)q1.z; f[7]  = (_Float16)q1.w;
  f[8]  = (_Float16)q2.x; f[9]  = (_Float16)q2.y; f[10] = (_Float16)q2.z; f[11] = (_Float16)q2.w;
  f[12] = (_Float16)q3.x; f[13] = (_Float16)q3.y; f[14] = (_Float16)q3.z; f[15] = (_Float16)q3.w;
  return f;
}

__device__ __forceinline__ v16h make_frag_nn(const float* W, int ldb, int col, int kb) {
  v16h f;
#pragma unroll
  for (int i = 0; i < 8; ++i) f[i]     = (_Float16)W[(size_t)(kb + i) * ldb + col];
#pragma unroll
  for (int i = 0; i < 8; ++i) f[8 + i] = (_Float16)W[(size_t)(kb + 16 + i) * ldb + col];
  return f;
}

// ---------------------------------------------------------------------------
// Generic WMMA GEMM: C[M,N] = act(A[M,K] @ B + bias) (+ resid)
//   BT=true : weights W are [N,K]  (torch Linear, x @ W^T)
//   BT=false: weights W are [K,N]  (moe einsum weights, gate)
// One wave (32 threads) computes a 64x64 tile via 4x4 v_wmma_f32_16x16x32_f16.
// M, N must be multiples of 64; K multiple of 32. A row stride = Kdim,
// C row stride = N, NN weight row stride = N.
// act: 0 = none, 1 = exact GELU.
// ---------------------------------------------------------------------------
template <bool BT>
__launch_bounds__(32)
__global__ void wmma_gemm_kernel(const float* __restrict__ A, const float* __restrict__ W,
                                 const float* __restrict__ bias, const float* resid,
                                 float* C, int Kdim, int N, int act) {
  const int lane = threadIdx.x & 31;
  const int half = lane >> 4;
  const int l    = lane & 15;
  const int m0   = blockIdx.y * 64;
  const int n0   = blockIdx.x * 64;

  v8f acc[4][4] = {};

  for (int k = 0; k < Kdim; k += 32) {
    const int kb = k + 8 * half;
    v16h af[4], bf[4];
#pragma unroll
    for (int i = 0; i < 4; ++i) {
      const float* ar = A + (size_t)(m0 + 16 * i + l) * Kdim;
      af[i] = make_frag(ar + kb, ar + kb + 16);
    }
#pragma unroll
    for (int j = 0; j < 4; ++j) {
      if (BT) {
        const float* wr = W + (size_t)(n0 + 16 * j + l) * Kdim;
        bf[j] = make_frag(wr + kb, wr + kb + 16);
      } else {
        bf[j] = make_frag_nn(W, N, n0 + 16 * j + l, kb);
      }
    }
#pragma unroll
    for (int i = 0; i < 4; ++i)
#pragma unroll
      for (int j = 0; j < 4; ++j)
        acc[i][j] = __builtin_amdgcn_wmma_f32_16x16x32_f16(
            false, af[i], false, bf[j], (short)0, acc[i][j], false, false);
  }

  // Epilogue. C/D layout: lane col = n0+16j+l, VGPR r -> row m0+16i + 8*half + r.
#pragma unroll
  for (int i = 0; i < 4; ++i) {
#pragma unroll
    for (int j = 0; j < 4; ++j) {
      const int col = n0 + 16 * j + l;
      const float bv = bias ? bias[col] : 0.f;
#pragma unroll
      for (int r = 0; r < 8; ++r) {
        const int row = m0 + 16 * i + 8 * half + r;
        float v = acc[i][j][r] + bv;
        if (act == 1) v = 0.5f * v * (1.f + erff(v * 0.70710678118654752440f));
        if (resid) v += resid[(size_t)row * N + col];
        C[(size_t)row * N + col] = v;
      }
    }
  }
}

// ---------------------------------------------------------------------------
// im2col for 16x16/stride-16 patch conv. Rows indexed as b*197 + n, n>=1 are
// patches (cls row n==0 and pad rows left as zero). col = ch*256 + ky*16 + kx.
// ---------------------------------------------------------------------------
__global__ void im2col_kernel(const float* __restrict__ x, float* __restrict__ col) {
  const int idx = blockIdx.x * 256 + threadIdx.x;
  if (idx >= MPAD * 768) return;
  const int r = idx / 768, c = idx % 768;
  const int b = r / 197, n = r % 197;
  float v = 0.f;
  if (b < 4 && n >= 1) {
    const int p = n - 1, py = p / 14, px = p % 14;
    const int ch = c >> 8, rem = c & 255, ky = rem >> 4, kx = rem & 15;
    v = x[(((size_t)b * 3 + ch) * 224 + (py * 16 + ky)) * 224 + (px * 16 + kx)];
  }
  col[idx] = v;
}

// h = concat(cls, patches) + pos_embed; pad rows zeroed.
__global__ void assemble_kernel(const float* __restrict__ pe, const float* __restrict__ cls,
                                const float* __restrict__ pos, float* __restrict__ h) {
  const int idx = blockIdx.x * 256 + threadIdx.x;
  if (idx >= MPAD * 384) return;
  const int r = idx / 384, c = idx % 384;
  const int n = r % 197;
  float v = 0.f;
  if (r < MREAL) v = ((n == 0) ? cls[c] : pe[idx]) + pos[(size_t)n * 384 + c];
  h[idx] = v;
}

// ---------------------------------------------------------------------------
// LayerNorm over 384 features, one wave per row (wave32: 12 elems/lane).
// ---------------------------------------------------------------------------
__global__ void layernorm_kernel(const float* __restrict__ x, const float* __restrict__ w,
                                 const float* __restrict__ b, float* __restrict__ y) {
  const int row  = blockIdx.x;
  const int lane = threadIdx.x;
  const float* xr = x + (size_t)row * 384;
  float v[12];
  float s = 0.f;
#pragma unroll
  for (int i = 0; i < 12; ++i) { v[i] = xr[lane + 32 * i]; s += v[i]; }
#pragma unroll
  for (int off = 16; off > 0; off >>= 1) s += __shfl_xor(s, off, 32);
  const float mean = s * (1.f / 384.f);
  float q = 0.f;
#pragma unroll
  for (int i = 0; i < 12; ++i) { const float d = v[i] - mean; q += d * d; }
#pragma unroll
  for (int off = 16; off > 0; off >>= 1) q += __shfl_xor(q, off, 32);
  const float inv = rsqrtf(q * (1.f / 384.f) + 1e-6f);
  float* yr = y + (size_t)row * 384;
#pragma unroll
  for (int i = 0; i < 12; ++i) {
    const int c = lane + 32 * i;
    yr[c] = (v[i] - mean) * inv * w[c] + b[c];
  }
}

// ---------------------------------------------------------------------------
// Attention: one block per (query token, head, batch). qkv rows are
// [q(384) | k(384) | v(384)], row stride 1152. 197x197 softmax in LDS.
// ---------------------------------------------------------------------------
__global__ void attention_kernel(const float* __restrict__ qkv, float* __restrict__ out) {
  const int n  = blockIdx.x;   // 0..196
  const int hh = blockIdx.y;   // 0..5
  const int b  = blockIdx.z;   // 0..3
  const int tid = threadIdx.x; // 0..255
  __shared__ float qs[64];
  __shared__ float p[197];
  __shared__ float red[256];
  const size_t rowq = (size_t)(b * 197 + n);
  if (tid < 64) qs[tid] = qkv[rowq * 1152 + hh * 64 + tid];
  __syncthreads();
  float s = -3.0e38f;
  if (tid < 197) {
    const float* kr = qkv + (size_t)(b * 197 + tid) * 1152 + 384 + hh * 64;
    float acc = 0.f;
#pragma unroll 8
    for (int d = 0; d < 64; ++d) acc += qs[d] * kr[d];
    s = acc * 0.125f;  // 1/sqrt(64)
  }
  red[tid] = s;
  __syncthreads();
  for (int off = 128; off > 0; off >>= 1) {
    if (tid < off) red[tid] = fmaxf(red[tid], red[tid + off]);
    __syncthreads();
  }
  const float mx = red[0];
  __syncthreads();
  float e = 0.f;
  if (tid < 197) { e = expf(s - mx); p[tid] = e; }
  red[tid] = e;
  __syncthreads();
  for (int off = 128; off > 0; off >>= 1) {
    if (tid < off) red[tid] += red[tid + off];
    __syncthreads();
  }
  const float inv = 1.f / red[0];
  if (tid < 64) {
    float acc = 0.f;
    for (int j = 0; j < 197; ++j)
      acc += p[j] * qkv[(size_t)(b * 197 + j) * 1152 + 768 + hh * 64 + tid];
    out[rowq * 384 + hh * 64 + tid] = acc * inv;
  }
}

// ---------------------------------------------------------------------------
// Noiseless top-2 gating: logits = y @ gw ([384,8]); softmax over top-2,
// dense gate row written (zeros elsewhere). Tie-break matches lax.top_k
// (lower index wins) via strict '>'.
// ---------------------------------------------------------------------------
__global__ void gate_topk_kernel(const float* __restrict__ y, const float* __restrict__ gw,
                                 float* __restrict__ G) {
  const int row  = blockIdx.x;
  const int lane = threadIdx.x;
  float logit = 0.f;
  if (lane < 8) {
    const float* yr = y + (size_t)row * 384;
    for (int d = 0; d < 384; ++d) logit += yr[d] * gw[d * 8 + lane];
  }
  float lv[8];
#pragma unroll
  for (int e = 0; e < 8; ++e) lv[e] = __shfl(logit, e, 32);
  int i0 = 0; float v0 = lv[0];
#pragma unroll
  for (int e = 1; e < 8; ++e) if (lv[e] > v0) { v0 = lv[e]; i0 = e; }
  int i1 = -1; float v1 = -3.0e38f;
#pragma unroll
  for (int e = 0; e < 8; ++e) if (e != i0 && lv[e] > v1) { v1 = lv[e]; i1 = e; }
  const float e1 = expf(v1 - v0);
  const float den = 1.f + e1;
  if (lane < 8)
    G[(size_t)row * 8 + lane] = (lane == i0) ? (1.f / den) : ((lane == i1) ? (e1 / den) : 0.f);
}

// h += sum_e G[row,e] * eo[e][row,:]
__global__ void moe_combine_kernel(float* __restrict__ h, const float* __restrict__ G,
                                   const float* __restrict__ eo) {
  const int idx = blockIdx.x * 256 + threadIdx.x;
  if (idx >= MPAD * 384) return;
  const int row = idx / 384;
  const int col = idx % 384;
  float acc = h[idx];
#pragma unroll
  for (int e = 0; e < 8; ++e)
    acc += G[(size_t)row * 8 + e] * eo[((size_t)e * MPAD + row) * 384 + col];
  h[idx] = acc;
}

// ---------------------------------------------------------------------------
extern "C" void kernel_launch(void* const* d_in, const int* in_sizes, int n_in,
                              void* d_out, int out_size, void* d_ws, size_t ws_size,
                              hipStream_t stream) {
  const float* x       = (const float*)d_in[0];
  const float* patch_w = (const float*)d_in[1];   // [384, 768] flat (NT)
  const float* patch_b = (const float*)d_in[2];
  const float* cls_tok = (const float*)d_in[3];
  const float* pos_emb = (const float*)d_in[4];
  const float* ln1_w   = (const float*)d_in[5];
  const float* ln1_b   = (const float*)d_in[6];
  const float* qkv_w   = (const float*)d_in[7];   // [12,1152,384] (NT)
  const float* qkv_b   = (const float*)d_in[8];
  const float* proj_w  = (const float*)d_in[9];   // [12,384,384] (NT)
  const float* proj_b  = (const float*)d_in[10];
  const float* ln2_w   = (const float*)d_in[11];
  const float* ln2_b   = (const float*)d_in[12];
  const float* fc1_w   = (const float*)d_in[13];  // [6,1536,384] (NT)
  const float* fc1_b   = (const float*)d_in[14];
  const float* fc2_w   = (const float*)d_in[15];  // [6,384,1536] (NT)
  const float* fc2_b   = (const float*)d_in[16];
  const float* gate_w  = (const float*)d_in[17];  // [6,384,8] (NN)
  const float* moe_w1  = (const float*)d_in[18];  // [6,8,384,1536] (NN)
  const float* moe_b1  = (const float*)d_in[19];
  const float* moe_w2  = (const float*)d_in[20];  // [6,8,1536,384] (NN)
  const float* moe_b2  = (const float*)d_in[21];
  const float* lnf_w   = (const float*)d_in[22];
  const float* lnf_b   = (const float*)d_in[23];
  (void)in_sizes; (void)n_in; (void)out_size;

  // Workspace carve-up (floats)
  float* ws = (float*)d_ws;
  float* h   = ws;                               // [MPAD, 384] residual stream
  float* x0  = h  + (size_t)MPAD * 384;          // [MPAD, 384] LN out / patch gemm out
  float* x1  = x0 + (size_t)MPAD * 384;          // [MPAD, 1536] qkv / fc hidden / im2col
  float* x2  = x1 + (size_t)MPAD * 1536;         // [MPAD, 384] attention out
  float* G   = x2 + (size_t)MPAD * 384;          // [MPAD, 8] gates
  float* eo  = G  + (size_t)MPAD * 8;            // [8, MPAD, 384] expert outputs
  const size_t total_floats = (size_t)MPAD * (384 + 384 + 1536 + 384 + 8) + (size_t)8 * MPAD * 384;
  if (ws_size < total_floats * sizeof(float)) return;

  hipMemsetAsync(d_ws, 0, total_floats * sizeof(float), stream);

  const dim3 blk32(32), blk256(256);
  const int MT = MPAD / 64;  // 13 M tiles
  const int elemblk = (MPAD * 384 + 255) / 256;

  // --- Patch embed: im2col -> GEMM (NT, K=768, N=384) -> assemble + pos ---
  im2col_kernel<<<(MPAD * 768 + 255) / 256, blk256, 0, stream>>>(x, x1);
  wmma_gemm_kernel<true><<<dim3(384 / 64, MT), blk32, 0, stream>>>(
      x1, patch_w, patch_b, nullptr, x0, 768, 384, 0);
  assemble_kernel<<<elemblk, blk256, 0, stream>>>(x0, cls_tok, pos_emb, h);

  // --- 12 transformer layers ---
  for (int i = 0; i < 12; ++i) {
    // Attention block
    layernorm_kernel<<<MPAD, blk32, 0, stream>>>(h, ln1_w + i * 384, ln1_b + i * 384, x0);
    wmma_gemm_kernel<true><<<dim3(1152 / 64, MT), blk32, 0, stream>>>(
        x0, qkv_w + (size_t)i * 1152 * 384, qkv_b + i * 1152, nullptr, x1, 384, 1152, 0);
    attention_kernel<<<dim3(197, 6, 4), blk256, 0, stream>>>(x1, x2);
    wmma_gemm_kernel<true><<<dim3(384 / 64, MT), blk32, 0, stream>>>(
        x2, proj_w + (size_t)i * 384 * 384, proj_b + i * 384, h, h, 384, 384, 0);

    // FFN block
    layernorm_kernel<<<MPAD, blk32, 0, stream>>>(h, ln2_w + i * 384, ln2_b + i * 384, x0);
    const int j = i / 2;
    if ((i & 1) == 0) {
      // Dense MLP: fc1 (GELU fused) then fc2 (+residual into h)
      wmma_gemm_kernel<true><<<dim3(1536 / 64, MT), blk32, 0, stream>>>(
          x0, fc1_w + (size_t)j * 1536 * 384, fc1_b + j * 1536, nullptr, x1, 384, 1536, 1);
      wmma_gemm_kernel<true><<<dim3(384 / 64, MT), blk32, 0, stream>>>(
          x1, fc2_w + (size_t)j * 384 * 1536, fc2_b + j * 384, h, h, 1536, 384, 0);
    } else {
      // MoE: top-2 gates, dense per-expert FFN (identical math to reference),
      // gate-weighted combine.
      gate_topk_kernel<<<MPAD, blk32, 0, stream>>>(x0, gate_w + (size_t)j * 384 * 8, G);
      for (int e = 0; e < 8; ++e) {
        const float* w1 = moe_w1 + ((size_t)j * 8 + e) * 384 * 1536;
        const float* b1 = moe_b1 + ((size_t)j * 8 + e) * 1536;
        const float* w2 = moe_w2 + ((size_t)j * 8 + e) * 1536 * 384;
        const float* b2 = moe_b2 + ((size_t)j * 8 + e) * 384;
        wmma_gemm_kernel<false><<<dim3(1536 / 64, MT), blk32, 0, stream>>>(
            x0, w1, b1, nullptr, x1, 384, 1536, 1);
        wmma_gemm_kernel<false><<<dim3(384 / 64, MT), blk32, 0, stream>>>(
            x1, w2, b2, nullptr, eo + (size_t)e * MPAD * 384, 1536, 384, 0);
      }
      moe_combine_kernel<<<elemblk, blk256, 0, stream>>>(h, G, eo);
    }
  }

  // --- Final LayerNorm straight into d_out (rows 0..787 == [4,197,384]) ---
  layernorm_kernel<<<MREAL, blk32, 0, stream>>>(h, lnf_w, lnf_b, (float*)d_out);
}
